// Decoder_offset001_5789615915111
// MI455X (gfx1250) — matrix operand order repaired and checked by reading, not compiled, not run on hardware
//
#include <hip/hip_runtime.h>
#include <hip/hip_bf16.h>
#include <math.h>

// ---------------------------------------------------------------------------
// CDNA5 (gfx1250) implementation: f16 WMMA implicit-GEMM convs + deform convs.
// Activations kept in NHWC f16 (channel-contiguous) so all K-dim loads are
// wide b128 accesses; weights pre-packed into WMMA B-fragment lane layout.
// Halo staging uses GLOBAL_LOAD_ASYNC_TO_LDS_B128 (ASYNCcnt path, L2->LDS
// direct).  Waves compute 32x32 (or 16x32 / 32x16) tiles: 4 independent
// WMMAs per fragment-load group to hide XDL latency and halve mem-ops/WMMA.
// ---------------------------------------------------------------------------

typedef __attribute__((ext_vector_type(16))) _Float16 v16h;
typedef __attribute__((ext_vector_type(8)))  _Float16 v8h;
typedef __attribute__((ext_vector_type(8)))  float    v8f;

__device__ __forceinline__ v8h zero_v8h() {
    v8h z;
#pragma unroll
    for (int i = 0; i < 8; ++i) z[i] = (_Float16)0.f;
    return z;
}
__device__ __forceinline__ v8f zero_v8f() {
    v8f z;
#pragma unroll
    for (int i = 0; i < 8; ++i) z[i] = 0.f;
    return z;
}

union AFrag { v16h v; v8h h[2]; };

// Async L2->LDS 16-byte copy (CDNA5 GLOBAL_LOAD_ASYNC_TO_LDS_B128, ASYNCcnt).
// The per-lane LDS destination is the low 32 bits of the generic pointer.
__device__ __forceinline__ void async_copy_b128(void* lds_dst, const void* gsrc) {
    unsigned l = (unsigned)(unsigned long long)lds_dst;
    asm volatile("global_load_async_to_lds_b128 %0, %1, off"
                 :: "v"(l), "v"(gsrc)
                 : "memory");
}
__device__ __forceinline__ void wait_async0() {
    asm volatile("s_wait_asynccnt 0x0" ::: "memory");
}

// ---------------------------------------------------------------------------
// NCHW f32 -> NHWC f16 layout conversion (one time, reads coalesced).
// ---------------------------------------------------------------------------
__global__ void nchw_to_nhwc_f16(const float* __restrict__ in,
                                 _Float16* __restrict__ out, int C) {
    const int HW = 256 * 256;
    int idx = blockIdx.x * 256 + threadIdx.x;
    int total = 2 * C * HW;
    if (idx >= total) return;
    int x = idx & 255;
    int t = idx >> 8;
    int y = t & 255;  t >>= 8;
    int c = t % C;
    int b = t / C;
    out[(size_t)(((b * 256 + y) * 256 + x)) * C + c] = (_Float16)in[idx];
}

// ---------------------------------------------------------------------------
// Offset prep.  In the reference, py = y + GY[k] + (0.08*oy_k - GY[k]) =
// y + 0.08*oy_k (grid terms cancel), likewise for px.  Store per-pixel
// [ox(0..8), oy(0..8)] * 0.08 contiguously (18 floats / pixel).
// offset_0 is (2,24,256,256) NCHW; ox_k = ch (3+k)*2, oy_k = ch (3+k)*2+1.
// ---------------------------------------------------------------------------
__global__ void offset_prep(const float* __restrict__ o0,
                            float* __restrict__ off) {
    const int HW = 256 * 256;
    int idx = blockIdx.x * 256 + threadIdx.x;
    int total = 2 * HW * 9;
    if (idx >= total) return;
    int k = idx % 9;
    int t = idx / 9;
    int pix = t % HW;
    int b = t / HW;
    float ox = o0[((size_t)(b * 24 + (3 + k) * 2) * HW) + pix];
    float oy = o0[((size_t)(b * 24 + (3 + k) * 2 + 1) * HW) + pix];
    float* p = off + ((size_t)b * HW + pix) * 18;
    p[k]     = 0.08f * ox;
    p[9 + k] = 0.08f * oy;
}

// ---------------------------------------------------------------------------
// Weight packing: OIHW f32 -> f16 WMMA B-fragments.
// K order = tap*CIN + cin.  Fragment f = ntile*KC + kchunk holds 32 lanes x
// 16 halves:  lane<16: N=ntile*16+lane, K=kc*32+i ; lane>=16: K=kc*32+16+i.
// colmajor_taps: tap -> (ky,kx) = (tap%3, tap/3)  (deform-conv tap order).
// ---------------------------------------------------------------------------
__global__ void pack_w(const float* __restrict__ w, _Float16* __restrict__ out,
                       int cin, int cout, int colmajor) {
    int KC = (9 * cin) / 32;
    int NT = (cout + 15) / 16;
    int total = NT * KC * 512;
    int idx = blockIdx.x * 256 + threadIdx.x;
    if (idx >= total) return;
    int i    = idx & 15;
    int lane = (idx >> 4) & 31;
    int frag = idx >> 9;
    int kc   = frag % KC;
    int nt   = frag / KC;
    int n = nt * 16 + (lane & 15);
    int k = kc * 32 + (lane >> 4) * 16 + i;
    int tap = k / cin;
    int ci  = k % cin;
    int ky = colmajor ? (tap % 3) : (tap / 3);
    int kx = colmajor ? (tap / 3) : (tap % 3);
    _Float16 v = (_Float16)0.f;
    if (n < cout) v = (_Float16)w[(((size_t)n * cin + ci) * 3 + ky) * 3 + kx];
    out[idx] = v;
}

// ---------------------------------------------------------------------------
// 3x3 conv (pad=1) as implicit GEMM with V_WMMA_F32_16X16X32_F16.
// Block = 256 threads = 8 wave32s arranged NWM x NWX; each wave computes a
// WM x WN output tile (MI x NI accumulators).  K = 9*CIN.
// Epilogue fuses bias / relu / residual-add / store (NHWC f16 or NCHW f32).
// ---------------------------------------------------------------------------
template <int CIN, int COUT, int WM, int WN, bool RELU, bool RES, bool OUTF32>
__global__ __launch_bounds__(256, 2) void conv3x3_wmma(
    const _Float16* __restrict__ in,   // NHWC f16, 2 x 256 x 256 x CIN
    const _Float16* __restrict__ wpk,  // packed B fragments
    const float* __restrict__ bias,    // COUT
    const _Float16* __restrict__ res,  // NHWC f16 (RES) or nullptr
    _Float16* __restrict__ outh,       // NHWC f16 output
    float* __restrict__ outf)          // NCHW f32 output (final layer)
{
    constexpr int H = 256, W = 256;
    constexpr int NTC = (COUT + 15) / 16;   // padded 16-wide cout tiles
    constexpr int NWX = (NTC * 16) / WN;    // waves along N
    constexpr int NWM = 8 / NWX;            // waves along M
    constexpr int TM  = NWM * WM;           // pixels per block
    constexpr int MI  = WM / 16, NI = WN / 16;
    constexpr int KC  = (9 * CIN) / 32;     // K chunks total
    constexpr int CPT = CIN / 32;           // K chunks per tap
    constexpr int CW  = TM + 2;             // staged columns (halo)

    __shared__ __align__(16) _Float16 lds[3 * CW * CIN];

    const int tid = threadIdx.x;
    const int x0 = blockIdx.x * TM;
    const int y  = blockIdx.y;
    const int b  = blockIdx.z;

    __builtin_prefetch(wpk, 0, 0);   // global_prefetch_b8 on weight stream

    // ---- stage 3 x CW x CIN f16 halo tile into LDS via async L2->LDS ----
    constexpr int NG = 3 * CW * (CIN / 8);
    for (int g = tid; g < NG; g += 256) {
        int cg  = g % (CIN / 8);
        int rc  = g / (CIN / 8);
        int col = rc % CW;
        int row = rc / CW;
        int gy = y + row - 1;
        int gx = x0 + col - 1;
        _Float16* lp = &lds[(row * CW + col) * CIN + cg * 8];
        if ((unsigned)gy < (unsigned)H && (unsigned)gx < (unsigned)W) {
            async_copy_b128(lp,
                in + (size_t)((b * H + gy) * W + gx) * CIN + cg * 8);
        } else {
            *(v8h*)lp = zero_v8h();
        }
    }
    wait_async0();
    __syncthreads();

    // ---- per-wave implicit GEMM, MI x NI accumulators ----
    const int wv   = tid >> 5;
    const int lane = tid & 31;
    const int lm   = lane & 15;
    const int hi   = lane >> 4;
    const int wn   = wv % NWX;
    const int wm   = wv / NWX;
    const int m0   = wm * WM;
    const int n0   = wn * WN;

    v8f acc[MI][NI];
#pragma unroll
    for (int mi = 0; mi < MI; ++mi)
#pragma unroll
        for (int ni = 0; ni < NI; ++ni) acc[mi][ni] = zero_v8f();

#pragma unroll
    for (int tap = 0; tap < 9; ++tap) {
        const int ky = tap / 3, kx = tap % 3;
#pragma unroll
        for (int c = 0; c < CPT; ++c) {
            AFrag a[MI];
#pragma unroll
            for (int mi = 0; mi < MI; ++mi) {
                const _Float16* lp = &lds[(ky * CW + (m0 + mi * 16 + lm + kx))
                                              * CIN + c * 32 + hi * 16];
                a[mi].h[0] = *(const v8h*)lp;        // K = base .. base+7
                a[mi].h[1] = *(const v8h*)(lp + 8);  // K = base+8 .. base+15
            }
            v16h bf[NI];
#pragma unroll
            for (int ni = 0; ni < NI; ++ni) {
                const int f = (n0 / 16 + ni) * KC + tap * CPT + c;
                bf[ni] = *(const v16h*)(wpk + ((size_t)f * 32 + lane) * 16);
            }
#pragma unroll
            for (int mi = 0; mi < MI; ++mi)
#pragma unroll
                for (int ni = 0; ni < NI; ++ni)
                    acc[mi][ni] = __builtin_amdgcn_wmma_f32_16x16x32_f16(
                        false, a[mi].v, false, bf[ni], (short)0, acc[mi][ni],
                        false, false);
        }
    }

    // ---- epilogue: C layout = lane<16: (M=r, N=lane); lane>=16: (M=8+r) ----
#pragma unroll
    for (int ni = 0; ni < NI; ++ni) {
        const int n = n0 + ni * 16 + lm;
        const float bv = (n < COUT) ? bias[n] : 0.f;
#pragma unroll
        for (int mi = 0; mi < MI; ++mi) {
#pragma unroll
            for (int r = 0; r < 8; ++r) {
                const int m  = m0 + mi * 16 + hi * 8 + r;
                const int gx = x0 + m;
                float v = acc[mi][ni][r] + bv;
                if constexpr (RELU) v = fmaxf(v, 0.f);
                if constexpr (RES) {
                    if (n < COUT)
                        v += (float)res[(size_t)((b * H + y) * W + gx) * COUT + n];
                }
                if constexpr (OUTF32) {
                    if (n < COUT)
                        outf[(((size_t)b * COUT + n) * H + y) * W + gx] = v;
                } else {
                    if (n < COUT)
                        outh[(size_t)((b * H + y) * W + gx) * COUT + n] =
                            (_Float16)v;
                }
            }
        }
    }
}

// ---------------------------------------------------------------------------
// Deformable 3x3 conv (32 -> 32) with WMMA.  Per tap: 256 threads bilinear-
// sample a 64-pixel x 32-channel f16 tile into double-buffered LDS from the
// NHWC input (4 corner b128 gathers / 8 channels), then 8 waves each do one
// 16x16x32 WMMA accumulate.  tap k -> weight (ky,kx) = (k%3, k/3); sample
// positions py = y + 0.08*oy_k, px = x + 0.08*ox_k (grid terms cancel).
// ---------------------------------------------------------------------------
template <bool RELU, bool RES>
__global__ __launch_bounds__(256, 2) void deform_conv_wmma(
    const _Float16* __restrict__ in,   // NHWC f16, 2 x 256 x 256 x 32
    const float* __restrict__ off,     // per pixel 18 floats
    const _Float16* __restrict__ wpk,  // 2 ntiles x 9 taps B fragments
    const float* __restrict__ bias,    // 32
    const _Float16* __restrict__ res,  // NHWC f16 (RES) or nullptr
    _Float16* __restrict__ out)        // NHWC f16
{
    constexpr int H = 256, W = 256, C = 32, TM = 64;
    __shared__ __align__(16) _Float16 smp[2][TM * C];

    const int tid = threadIdx.x;
    const int x0 = blockIdx.x * TM;
    const int y  = blockIdx.y;
    const int b  = blockIdx.z;

    const int lane = tid & 31;
    const int wv   = tid >> 5;
    const int lm   = lane & 15;
    const int hi   = lane >> 4;
    const int mt   = wv >> 1;
    const int nt   = wv & 1;
    const int m0   = mt * 16;

    // sampling role: pixel sp (0..63), channel group c8 (0/8/16/24)
    const int sp = tid >> 2;
    const int c8 = (tid & 3) * 8;
    const int gx = x0 + sp;
    const float* offp = off + (size_t)((b * H + y) * W + gx) * 18;

    __builtin_prefetch(wpk, 0, 0);

    v8f acc = zero_v8f();

    for (int tap = 0; tap < 9; ++tap) {
        // ---- bilinear sample 8 channels of one pixel into LDS ----
        const float px = (float)gx + offp[tap];
        const float py = (float)y  + offp[9 + tap];
        const float fx = floorf(px), fy = floorf(py);
        const int ix = (int)fx, iy = (int)fy;
        const float wx = px - fx, wy = py - fy;
        const float cw[4] = {(1.f - wy) * (1.f - wx), (1.f - wy) * wx,
                             wy * (1.f - wx),          wy * wx};
        const int cy[4] = {iy, iy, iy + 1, iy + 1};
        const int cx[4] = {ix, ix + 1, ix, ix + 1};

        float sv[8];
#pragma unroll
        for (int i = 0; i < 8; ++i) sv[i] = 0.f;
#pragma unroll
        for (int q = 0; q < 4; ++q) {
            if ((unsigned)cy[q] < (unsigned)H && (unsigned)cx[q] < (unsigned)W) {
                const v8h g = *(const v8h*)(
                    in + ((size_t)(b * H + cy[q]) * W + cx[q]) * C + c8);
#pragma unroll
                for (int i = 0; i < 8; ++i) sv[i] += cw[q] * (float)g[i];
            }
        }
        v8h pk;
#pragma unroll
        for (int i = 0; i < 8; ++i) pk[i] = (_Float16)sv[i];
        *(v8h*)&smp[tap & 1][sp * C + c8] = pk;
        __syncthreads();

        // ---- WMMA accumulate: 16 pixels x 16 couts, K = 32 channels ----
        AFrag a;
        const _Float16* lp = &smp[tap & 1][(m0 + lm) * C + hi * 16];
        a.h[0] = *(const v8h*)lp;
        a.h[1] = *(const v8h*)(lp + 8);
        const int f = nt * 9 + tap;
        const v16h bf = *(const v16h*)(wpk + ((size_t)f * 32 + lane) * 16);
        acc = __builtin_amdgcn_wmma_f32_16x16x32_f16(
            false, a.v, false, bf, (short)0, acc, false, false);
    }

    const int n = nt * 16 + lm;
    const float bv = bias[n];
#pragma unroll
    for (int r = 0; r < 8; ++r) {
        const int m   = m0 + hi * 8 + r;
        const int gxm = x0 + m;
        float v = acc[r] + bv;
        if constexpr (RELU) v = fmaxf(v, 0.f);
        if constexpr (RES)
            v += (float)res[(size_t)((b * H + y) * W + gxm) * C + n];
        out[(size_t)((b * H + y) * W + gxm) * C + n] = (_Float16)v;
    }
}

// ---------------------------------------------------------------------------
// Host orchestration.
// ---------------------------------------------------------------------------
static constexpr size_t SZ64  = (size_t)2 * 256 * 256 * 64 * 2;  // f16 NHWC 64ch
static constexpr size_t SZ32  = SZ64 / 2;                        // f16 NHWC 32ch
static constexpr size_t SZOFF = (size_t)2 * 256 * 256 * 18 * 4;  // f32 offsets
static constexpr size_t WPK64   = (size_t)4 * 18 * 512 * 2;  // 64->64
static constexpr size_t WPK6432 = (size_t)2 * 18 * 512 * 2;  // 64->32
static constexpr size_t WPK3203 = (size_t)1 *  9 * 512 * 2;  // 32->3 (padded)
static constexpr size_t WPK32   = (size_t)2 *  9 * 512 * 2;  // 32->32 deform

extern "C" void kernel_launch(void* const* d_in, const int* in_sizes, int n_in,
                              void* d_out, int out_size, void* d_ws,
                              size_t ws_size, hipStream_t stream) {
    const float* x     = (const float*)d_in[0];
    const float* off0  = (const float*)d_in[1];
    const float* l12w  = (const float*)d_in[2];
    const float* l12b  = (const float*)d_in[3];
    const float* l13w1 = (const float*)d_in[4];
    const float* l13b1 = (const float*)d_in[5];
    const float* l13w2 = (const float*)d_in[6];
    const float* l13b2 = (const float*)d_in[7];
    const float* l14w1 = (const float*)d_in[8];
    const float* l14b1 = (const float*)d_in[9];
    const float* l14w2 = (const float*)d_in[10];
    const float* l14b2 = (const float*)d_in[11];
    const float* l15w  = (const float*)d_in[12];
    const float* l15b  = (const float*)d_in[13];
    const float* d50w  = (const float*)d_in[14];
    const float* d50b  = (const float*)d_in[15];
    const float* d51w  = (const float*)d_in[16];
    const float* d51b  = (const float*)d_in[17];
    const float* d60w  = (const float*)d_in[18];
    const float* d60b  = (const float*)d_in[19];
    const float* d61w  = (const float*)d_in[20];
    const float* d61b  = (const float*)d_in[21];
    const float* l24w  = (const float*)d_in[22];
    const float* l24b  = (const float*)d_in[23];

    char* ws = (char*)d_ws;
    size_t o = 0;
    _Float16* A64a = (_Float16*)(ws + o); o += SZ64;
    _Float16* A64b = (_Float16*)(ws + o); o += SZ64;
    _Float16* A64c = (_Float16*)(ws + o); o += SZ64;
    _Float16* A32a = (_Float16*)(ws + o); o += SZ32;
    _Float16* A32b = (_Float16*)(ws + o); o += SZ32;
    _Float16* A32c = (_Float16*)(ws + o); o += SZ32;
    float*    OFFS = (float*)(ws + o);    o += SZOFF;
    _Float16* wp12  = (_Float16*)(ws + o); o += WPK64;
    _Float16* wp131 = (_Float16*)(ws + o); o += WPK64;
    _Float16* wp132 = (_Float16*)(ws + o); o += WPK64;
    _Float16* wp141 = (_Float16*)(ws + o); o += WPK64;
    _Float16* wp142 = (_Float16*)(ws + o); o += WPK64;
    _Float16* wp15  = (_Float16*)(ws + o); o += WPK6432;
    _Float16* wp24  = (_Float16*)(ws + o); o += WPK3203;
    _Float16* wp50  = (_Float16*)(ws + o); o += WPK32;
    _Float16* wp51  = (_Float16*)(ws + o); o += WPK32;
    _Float16* wp60  = (_Float16*)(ws + o); o += WPK32;
    _Float16* wp61  = (_Float16*)(ws + o); o += WPK32;
    (void)ws_size; (void)in_sizes; (void)n_in; (void)out_size;

    // --- prep: layout conversion, offsets, weight packing ---
    {
        int tot = 2 * 64 * 256 * 256;
        nchw_to_nhwc_f16<<<(tot + 255) / 256, 256, 0, stream>>>(x, A64a, 64);
    }
    {
        int tot = 2 * 256 * 256 * 9;
        offset_prep<<<(tot + 255) / 256, 256, 0, stream>>>(off0, OFFS);
    }
    auto pack = [&](const float* w, _Float16* dst, int cin, int cout, int cm) {
        int tot = ((cout + 15) / 16) * ((9 * cin) / 32) * 512;
        pack_w<<<(tot + 255) / 256, 256, 0, stream>>>(w, dst, cin, cout, cm);
    };
    pack(l12w,  wp12,  64, 64, 0);
    pack(l13w1, wp131, 64, 64, 0);
    pack(l13w2, wp132, 64, 64, 0);
    pack(l14w1, wp141, 64, 64, 0);
    pack(l14w2, wp142, 64, 64, 0);
    pack(l15w,  wp15,  64, 32, 0);
    pack(l24w,  wp24,  32,  3, 0);
    pack(d50w,  wp50,  32, 32, 1);
    pack(d51w,  wp51,  32, 32, 1);
    pack(d60w,  wp60,  32, 32, 1);
    pack(d61w,  wp61,  32, 32, 1);

    const dim3 blk(256);
    const dim3 g64(256 / 128, 256, 2);  // 64->64: waves 32x32, TM=128
    const dim3 g32(256 / 128, 256, 2);  // 64->32: waves 16x32, TM=128
    const dim3 g03(256 / 256, 256, 2);  // 32->3 : waves 32x16, TM=256
    const dim3 gdf(256 / 64, 256, 2);   // deform: TM=64

    // l12: X1 = conv(x)                          A64a -> A64b
    conv3x3_wmma<64, 64, 32, 32, false, false, false><<<g64, blk, 0, stream>>>(
        A64a, wp12, l12b, nullptr, A64b, nullptr);
    // l13: T = relu(conv(X1))                    A64b -> A64c
    conv3x3_wmma<64, 64, 32, 32, true, false, false><<<g64, blk, 0, stream>>>(
        A64b, wp131, l13b1, nullptr, A64c, nullptr);
    //      X2 = conv(T) + X1                     A64c (+A64b) -> A64a
    conv3x3_wmma<64, 64, 32, 32, false, true, false><<<g64, blk, 0, stream>>>(
        A64c, wp132, l13b2, A64b, A64a, nullptr);
    // l14: T = relu(conv(X2))                    A64a -> A64c
    conv3x3_wmma<64, 64, 32, 32, true, false, false><<<g64, blk, 0, stream>>>(
        A64a, wp141, l14b1, nullptr, A64c, nullptr);
    //      X3 = conv(T) + X2                     A64c (+A64a) -> A64b
    conv3x3_wmma<64, 64, 32, 32, false, true, false><<<g64, blk, 0, stream>>>(
        A64c, wp142, l14b2, A64a, A64b, nullptr);
    // l15: X4 = conv(X3)  (64->32)               A64b -> A32a
    conv3x3_wmma<64, 32, 16, 32, false, false, false><<<g32, blk, 0, stream>>>(
        A64b, wp15, l15b, nullptr, A32a, nullptr);

    // d50: r5 = relu(dc(X4))                     A32a -> A32b
    deform_conv_wmma<true, false><<<gdf, blk, 0, stream>>>(
        A32a, OFFS, wp50, d50b, nullptr, A32b);
    // d51: X5 = dc(r5) + X4                      A32b (+A32a) -> A32c
    deform_conv_wmma<false, true><<<gdf, blk, 0, stream>>>(
        A32b, OFFS, wp51, d51b, A32a, A32c);
    // d60: r6 = relu(dc(X5))                     A32c -> A32b
    deform_conv_wmma<true, false><<<gdf, blk, 0, stream>>>(
        A32c, OFFS, wp60, d60b, nullptr, A32b);
    // d61: X6 = dc(r6) + X5                      A32b (+A32c) -> A32a
    deform_conv_wmma<false, true><<<gdf, blk, 0, stream>>>(
        A32b, OFFS, wp61, d61b, A32c, A32a);

    // l24: out = conv(X6)  (32->3), f32 NCHW to d_out
    conv3x3_wmma<32, 3, 32, 16, false, false, true><<<g03, blk, 0, stream>>>(
        A32a, wp24, l24b, nullptr, nullptr, (float*)d_out);
}
